// MoELayer_35579509080438
// MI455X (gfx1250) — compile-verified, matrix operand order checked by reading, not run on hardware
//
#include <hip/hip_runtime.h>
#include <hip/hip_bf16.h>
#include <math.h>

// ---- problem constants (match reference) ----
#define NE     16     // experts
#define DMODEL 2048
#define FH     1408
#define NTOK   8192   // B*S = 4*2048
#define MT     32     // tokens per block tile (2 x 16-row WMMA halves)

// LDS strides (elements); row byte stride must be 16B aligned
#define XS 2056       // 2048 + 8
#define HS 1416       // 1408 + 8

typedef __attribute__((ext_vector_type(16))) __bf16 v16bf;
typedef __attribute__((ext_vector_type(8)))  float  v8f;

// ------------------------------------------------------------------
// Fragment loaders.  CDNA5 ISA 7.12.2, 16-bit A (16x32) layout:
//   lanes 0-15 : M = lane,   VGPR0..3 = K 0..7,  VGPR4..7 = K 16..23
//   lanes16-31 : M = lane-16,VGPR0..3 = K 8..15, VGPR4..7 = K 24..31
// B (32x16) mirrors with N in place of M, so lane n supplies row n of
// W (B = W^T).  Per lane: two contiguous 8-element chunks.
// ------------------------------------------------------------------
__device__ __forceinline__ v16bf frag_lds(const __bf16* base, int stride,
                                          int kb, int lane) {
  union { v16bf v; uint4 q[2]; } u;
  int m    = lane & 15;
  int koff = (lane & 16) ? 8 : 0;
  const __bf16* p0 = base + m * stride + kb + koff;
  u.q[0] = *(const uint4*)(p0);        // K = kb+koff    .. +7
  u.q[1] = *(const uint4*)(p0 + 16);   // K = kb+16+koff .. +7
  return u.v;
}

__device__ __forceinline__ v16bf frag_w_global(const float* __restrict__ W,
                                               int rowlen, int nb, int kb,
                                               int lane) {
  int n    = lane & 15;
  int koff = (lane & 16) ? 8 : 0;
  const float* p = W + (long long)(nb + n) * rowlen + kb + koff;
  float4 a0 = *(const float4*)(p);
  float4 a1 = *(const float4*)(p + 4);
  float4 b0 = *(const float4*)(p + 16);
  float4 b1 = *(const float4*)(p + 20);
  union { v16bf v; __bf16 h[16]; } u;
  u.h[0]=(__bf16)a0.x;  u.h[1]=(__bf16)a0.y;  u.h[2]=(__bf16)a0.z;  u.h[3]=(__bf16)a0.w;
  u.h[4]=(__bf16)a1.x;  u.h[5]=(__bf16)a1.y;  u.h[6]=(__bf16)a1.z;  u.h[7]=(__bf16)a1.w;
  u.h[8]=(__bf16)b0.x;  u.h[9]=(__bf16)b0.y;  u.h[10]=(__bf16)b0.z; u.h[11]=(__bf16)b0.w;
  u.h[12]=(__bf16)b1.x; u.h[13]=(__bf16)b1.y; u.h[14]=(__bf16)b1.z; u.h[15]=(__bf16)b1.w;
  return u.v;
}

// fast silu(g)*u: one v_rcp_f32 instead of the IEEE divide chain
__device__ __forceinline__ float silu_mul(float g, float u) {
  return g * __builtin_amdgcn_rcpf(1.0f + __expf(-g)) * u;
}

// ------------------------------------------------------------------
// Kernel 0: zero output + expert counts (d_out/d_ws are poisoned).
// ------------------------------------------------------------------
__global__ void moe_zero(float* __restrict__ out, long long n,
                         int* __restrict__ counts) {
  long long i = (long long)blockIdx.x * blockDim.x + threadIdx.x;
  long long stride = (long long)gridDim.x * blockDim.x;
  for (; i < n; i += stride) out[i] = 0.0f;
  if (blockIdx.x == 0 && threadIdx.x < NE) counts[threadIdx.x] = 0;
}

// ------------------------------------------------------------------
// Kernel 1: router.  One wave32 per token.
// ------------------------------------------------------------------
__global__ void moe_router(const float* __restrict__ x,
                           const float* __restrict__ gate_w,
                           const float* __restrict__ ebias,
                           float* __restrict__ combine,
                           int* __restrict__ counts,
                           int* __restrict__ lists) {
  int wave = threadIdx.x >> 5;
  int lane = threadIdx.x & 31;
  int token = blockIdx.x * 8 + wave;
  if (token >= NTOK) return;

  const float* xr = x + (long long)token * DMODEL;
  float acc[NE];
#pragma unroll
  for (int e = 0; e < NE; ++e) acc[e] = 0.0f;

  for (int d = lane; d < DMODEL; d += 32) {
    float xv = xr[d];
#pragma unroll
    for (int e = 0; e < NE; ++e) acc[e] += xv * gate_w[e * DMODEL + d];
  }
#pragma unroll
  for (int e = 0; e < NE; ++e) {
#pragma unroll
    for (int off = 16; off > 0; off >>= 1)
      acc[e] += __shfl_xor(acc[e], off, 32);
  }

  if (lane == 0) {
    float mx = acc[0];
#pragma unroll
    for (int e = 1; e < NE; ++e) mx = fmaxf(mx, acc[e]);
    float pe[NE];
#pragma unroll
    for (int e = 0; e < NE; ++e) pe[e] = __expf(acc[e] - mx);

    float biased[NE];
#pragma unroll
    for (int e = 0; e < NE; ++e) biased[e] = acc[e] + ebias[e];
    int i0 = 0;
#pragma unroll
    for (int e = 1; e < NE; ++e) if (biased[e] > biased[i0]) i0 = e;
    int i1 = (i0 == 0) ? 1 : 0;
#pragma unroll
    for (int e = 0; e < NE; ++e)
      if (e != i0 && biased[e] > biased[i1]) i1 = e;

    float denom = pe[i0] + pe[i1];
    float w0 = pe[i0] / denom, w1 = pe[i1] / denom;

    float* crow = combine + (long long)token * NE;
#pragma unroll
    for (int e = 0; e < NE; ++e) crow[e] = 0.0f;
    crow[i0] = w0;
    crow[i1] = w1;

    int s0 = atomicAdd(&counts[i0], 1);
    lists[i0 * NTOK + s0] = token;
    int s1 = atomicAdd(&counts[i1], 1);
    lists[i1 * NTOK + s1] = token;
  }
}

// ------------------------------------------------------------------
// Kernel 2: grouped swiglu.  Block = (expert, 32-token tile), 8 waves.
// Each weight fragment (B) feeds TWO 16-row WMMAs (M halves), halving
// weight traffic per token vs a 16-token tile.  K-loops unrolled 2x so
// the scheduler can overlap next iteration's global weight loads with
// the current iteration's WMMAs.
// Expert id 16 == shared expert (all tokens, weight 1).
// ------------------------------------------------------------------
__launch_bounds__(256, 1)
__global__ void moe_gemm(const float* __restrict__ x,
                         const float* __restrict__ w_gate,
                         const float* __restrict__ w_up,
                         const float* __restrict__ w_down,
                         const float* __restrict__ sw_gate,
                         const float* __restrict__ sw_up,
                         const float* __restrict__ sw_down,
                         const float* __restrict__ combine,
                         const int* __restrict__ counts,
                         const int* __restrict__ lists,
                         float* __restrict__ out) {
  __shared__ __bf16 Xs[MT * XS];   // 32 x D (bf16)  ~131.6 KB
  __shared__ __bf16 Hs[MT * HS];   // 32 x F (bf16)  ~90.6 KB
  __shared__ float  wts[MT];
  __shared__ int    toks[MT];

  const int e = blockIdx.y;                       // 0..15 routed, 16 shared
  const int nTok = (e == NE) ? NTOK : counts[e];
  const int base = blockIdx.x * MT;
  if (base >= nTok) return;

  const float* Wg = (e == NE) ? sw_gate : w_gate + (size_t)e * FH * DMODEL;
  const float* Wu = (e == NE) ? sw_up   : w_up   + (size_t)e * FH * DMODEL;
  const float* Wd = (e == NE) ? sw_down : w_down + (size_t)e * DMODEL * FH;

  const int tid = threadIdx.x;

  if (tid < MT) {
    int idx = base + tid;
    int tok = 0;
    float w = 0.0f;
    if (idx < nTok) {
      tok = (e == NE) ? idx : lists[e * NTOK + idx];
      w   = (e == NE) ? 1.0f : combine[(long long)tok * NE + e];
    }
    toks[tid] = tok;
    wts[tid]  = w;
  }
  __syncthreads();

  // stage X tile: fp32 -> bf16 into LDS. 8 threads/row, 256 elems each.
  {
    int row = tid >> 3, chunk = tid & 7;
    const float* src = x + (long long)toks[row] * DMODEL + chunk * 256;
    __bf16* dst = Xs + row * XS + chunk * 256;
#pragma unroll
    for (int i = 0; i < 256; i += 4) {
      float4 v = *(const float4*)(src + i);
      dst[i + 0] = (__bf16)v.x; dst[i + 1] = (__bf16)v.y;
      dst[i + 2] = (__bf16)v.z; dst[i + 3] = (__bf16)v.w;
    }
  }
  __syncthreads();

  const int wave = tid >> 5, lane = tid & 31;
  const int hi = (lane >> 4) & 1, nn = lane & 15;

  // ---- Phase A: 88 F-tiles, wave w handles ft = w + 8*r; 2 M-halves ----
  for (int r = 0; r < 11; ++r) {
    int nb = (wave + 8 * r) * 16;
    v8f g0 = {}, u0 = {}, g1 = {}, u1 = {};
#pragma unroll 2
    for (int kb = 0; kb < DMODEL; kb += 32) {
      __builtin_prefetch(Wg + (long long)nb * DMODEL + kb + 64, 0, 0);
      __builtin_prefetch(Wu + (long long)nb * DMODEL + kb + 64, 0, 0);
      v16bf a0 = frag_lds(Xs,           XS, kb, lane);   // tokens 0..15
      v16bf a1 = frag_lds(Xs + 16 * XS, XS, kb, lane);   // tokens 16..31
      v16bf bg = frag_w_global(Wg, DMODEL, nb, kb, lane);
      v16bf bu = frag_w_global(Wu, DMODEL, nb, kb, lane);
      g0 = __builtin_amdgcn_wmma_f32_16x16x32_bf16(false, a0, false, bg,
                                                   (short)0, g0, false, false);
      g1 = __builtin_amdgcn_wmma_f32_16x16x32_bf16(false, a1, false, bg,
                                                   (short)0, g1, false, false);
      u0 = __builtin_amdgcn_wmma_f32_16x16x32_bf16(false, a0, false, bu,
                                                   (short)0, u0, false, false);
      u1 = __builtin_amdgcn_wmma_f32_16x16x32_bf16(false, a1, false, bu,
                                                   (short)0, u1, false, false);
    }
#pragma unroll
    for (int rr = 0; rr < 8; ++rr) {      // C/D layout: row = rr + 8*hi
      int m = rr + 8 * hi;
      Hs[m * HS + nb + nn]        = (__bf16)silu_mul(g0[rr], u0[rr]);
      Hs[(m + 16) * HS + nb + nn] = (__bf16)silu_mul(g1[rr], u1[rr]);
    }
  }
  __syncthreads();

  // ---- Phase B: 128 D-tiles, wave w handles dt = w*16 + t; 2 M-halves ----
  for (int t = 0; t < 16; ++t) {
    int nb = (wave * 16 + t) * 16;
    v8f c0 = {}, c1 = {};
#pragma unroll 2
    for (int kb = 0; kb < FH; kb += 32) {
      __builtin_prefetch(Wd + (long long)nb * FH + kb + 64, 0, 0);
      v16bf a0 = frag_lds(Hs,           HS, kb, lane);
      v16bf a1 = frag_lds(Hs + 16 * HS, HS, kb, lane);
      v16bf b  = frag_w_global(Wd, FH, nb, kb, lane);
      c0 = __builtin_amdgcn_wmma_f32_16x16x32_bf16(false, a0, false, b,
                                                   (short)0, c0, false, false);
      c1 = __builtin_amdgcn_wmma_f32_16x16x32_bf16(false, a1, false, b,
                                                   (short)0, c1, false, false);
    }
#pragma unroll
    for (int rr = 0; rr < 8; ++rr) {
      int m = rr + 8 * hi;
      float w0 = wts[m];
      if (w0 != 0.0f)
        unsafeAtomicAdd(out + (long long)toks[m] * DMODEL + nb + nn,
                        c0[rr] * w0);
      float w1 = wts[m + 16];
      if (w1 != 0.0f)
        unsafeAtomicAdd(out + (long long)toks[m + 16] * DMODEL + nb + nn,
                        c1[rr] * w1);
    }
  }
}

// ------------------------------------------------------------------
extern "C" void kernel_launch(void* const* d_in, const int* in_sizes, int n_in,
                              void* d_out, int out_size, void* d_ws, size_t ws_size,
                              hipStream_t stream) {
  const float* x       = (const float*)d_in[0];
  const float* gate_w  = (const float*)d_in[1];
  const float* w_gate  = (const float*)d_in[2];
  const float* w_up    = (const float*)d_in[3];
  const float* w_down  = (const float*)d_in[4];
  const float* sw_gate = (const float*)d_in[5];
  const float* sw_up   = (const float*)d_in[6];
  const float* sw_down = (const float*)d_in[7];
  const float* ebias   = (const float*)d_in[8];
  float* out = (float*)d_out;

  // workspace layout
  int*   counts  = (int*)d_ws;                                  // 16 ints
  int*   lists   = (int*)((char*)d_ws + 256);                   // 16*8192 ints
  float* combine = (float*)((char*)d_ws + 256 + NE * NTOK * 4); // 8192*16 f32

  long long outN = (long long)NTOK * DMODEL;
  moe_zero<<<4096, 256, 0, stream>>>(out, outN, counts);
  moe_router<<<NTOK / 8, 256, 0, stream>>>(x, gate_w, ebias, combine, counts,
                                           lists);
  moe_gemm<<<dim3(NTOK / MT, NE + 1), 256, 0, stream>>>(
      x, w_gate, w_up, w_down, sw_gate, sw_up, sw_down, combine, counts, lists,
      out);
}
